// H2GFormerLayer_59253368816028
// MI455X (gfx1250) — matrix operand order repaired and checked by reading, not compiled, not run on hardware
//
#include <hip/hip_runtime.h>
#include <hip/hip_bf16.h>

typedef __attribute__((ext_vector_type(16))) _Float16 v16h;
typedef __attribute__((ext_vector_type(8)))  float    v8f;

#define NN 50000
#define EE 300000
#define DD 128
#define HH 8
#define MROWS 80   // rows per block (5 row-tiles per wave); 50000/80 = 625 exact

// ---------------------------------------------------------------- utilities
__global__ void fill_f32(float* p, float v, int n) {
  int i = blockIdx.x * blockDim.x + threadIdx.x;
  if (i < n) p[i] = v;
}
__global__ void copy_f32(const float* __restrict__ a, float* __restrict__ b, int n) {
  int i = blockIdx.x * blockDim.x + threadIdx.x;
  if (i < n) b[i] = a[i];
}
__global__ void conv_f32_f16(const float* __restrict__ a, _Float16* __restrict__ b, int n) {
  int i = blockIdx.x * blockDim.x + threadIdx.x;
  if (i < n) b[i] = (_Float16)a[i];
}
// transpose-convert: a is K x Nc row-major f32, b is Nc x K row-major f16
__global__ void convT_f32_f16(const float* __restrict__ a, _Float16* __restrict__ b,
                              int K, int Nc) {
  int i = blockIdx.x * blockDim.x + threadIdx.x;
  if (i >= K * Nc) return;
  int c = i / K, k = i - c * K;
  b[i] = (_Float16)a[k * Nc + c];
}

__device__ inline void atomicMaxF(float* addr, float val) {
  unsigned int* ua = (unsigned int*)addr;
  unsigned int old = *ua;
  while (__uint_as_float(old) < val) {
    unsigned int assumed = old;
    old = atomicCAS(ua, assumed, __float_as_uint(val));
    if (old == assumed) break;
  }
}

// ---------------------------------------------------------------- layernorm
// One wave (32 lanes) per row of 128 floats; float4 per lane.
__global__ void ln_kernel(const float* __restrict__ x, const float* __restrict__ g,
                          const float* __restrict__ b, float* __restrict__ of,
                          _Float16* __restrict__ oh) {
  int wid = threadIdx.x >> 5, lane = threadIdx.x & 31;
  int row = blockIdx.x * 8 + wid;
  if (row >= NN) return;
  float4 v = *(const float4*)(x + (size_t)row * DD + lane * 4);
  float s = v.x + v.y + v.z + v.w;
#pragma unroll
  for (int m2 = 16; m2 >= 1; m2 >>= 1) s += __shfl_xor(s, m2);
  float mu = s * (1.0f / DD);
  float d0 = v.x - mu, d1 = v.y - mu, d2 = v.z - mu, d3 = v.w - mu;
  float ss = d0 * d0 + d1 * d1 + d2 * d2 + d3 * d3;
#pragma unroll
  for (int m2 = 16; m2 >= 1; m2 >>= 1) ss += __shfl_xor(ss, m2);
  float rstd = rsqrtf(ss * (1.0f / DD) + 1e-5f);
  int c0 = lane * 4;
  float y0 = d0 * rstd * g[c0 + 0] + b[c0 + 0];
  float y1 = d1 * rstd * g[c0 + 1] + b[c0 + 1];
  float y2 = d2 * rstd * g[c0 + 2] + b[c0 + 2];
  float y3 = d3 * rstd * g[c0 + 3] + b[c0 + 3];
  if (of) {
    float4 r; r.x = y0; r.y = y1; r.z = y2; r.w = y3;
    *(float4*)(of + (size_t)row * DD + c0) = r;
  }
  if (oh) {
    _Float16* p = oh + (size_t)row * DD + c0;
    p[0] = (_Float16)y0; p[1] = (_Float16)y1; p[2] = (_Float16)y2; p[3] = (_Float16)y3;
  }
}

// ---------------------------------------------------------------- degrees / GCN
__global__ void deg_accum(const int* __restrict__ src, const int* __restrict__ dst,
                          float* __restrict__ dout, float* __restrict__ din) {
  int e = blockIdx.x * blockDim.x + threadIdx.x;
  if (e >= EE) return;
  atomicAdd(&dout[src[e]], 1.0f);
  atomicAdd(&din[dst[e]], 1.0f);
}
__global__ void rs_finalize(float* p, int n) {
  int i = blockIdx.x * blockDim.x + threadIdx.x;
  if (i < n) p[i] = rsqrtf(fmaxf(p[i], 1.0f));
}
// wave per edge; lane handles 4 contiguous floats
__global__ void gcn_scatter(const float* __restrict__ hl, const int* __restrict__ src,
                            const int* __restrict__ dst, const float* __restrict__ rs_out,
                            float* __restrict__ agg) {
  int t = blockIdx.x * blockDim.x + threadIdx.x;
  int e = t >> 5;
  if (e >= EE) return;
  int lane = t & 31;
  int s = src[e], d = dst[e];
  float rs = rs_out[s];
  float4 v = *(const float4*)(hl + (size_t)s * DD + lane * 4);
  float* ap = agg + (size_t)d * DD + lane * 4;
  atomicAdd(ap + 0, v.x * rs);
  atomicAdd(ap + 1, v.y * rs);
  atomicAdd(ap + 2, v.z * rs);
  atomicAdd(ap + 3, v.w * rs);
}
__global__ void scale_rows_f16(const float* __restrict__ agg, const float* __restrict__ rs_in,
                               _Float16* __restrict__ xh) {
  int i = blockIdx.x * blockDim.x + threadIdx.x;
  if (i >= NN * DD) return;
  xh[i] = (_Float16)(agg[i] * rs_in[i >> 7]);
}

// ---------------------------------------------------------------- WMMA GEMM
// C[N x Nc] = A[N x K](f16) @ W(transposed: Nc x K, f16) + bias, epilogue by mode:
//   mode 0: Cacc += result (f32)
//   mode 1: Oh = (f16)result
//   mode 2: Oh = (f16)gelu_exact(result)
// Block: 256 threads = 8 waves. Block covers MROWS x 128 outputs; each wave owns one
// 16-column tile and 5 stacked 16-row tiles (B fragment reused across 5 WMMAs).
__global__ void gemm_wmma_kernel(const _Float16* __restrict__ A, int K,
                                 const _Float16* __restrict__ Wt, int NcTot,
                                 const float* __restrict__ bias,
                                 float* __restrict__ Cacc,
                                 _Float16* __restrict__ Oh, int mode) {
  extern __shared__ _Float16 smem[];
  _Float16* Ah = smem;                 // MROWS x 128 chunk, row-major
  _Float16* Wcm = smem + MROWS * 128;  // 128 cols x 128 k, column-major (k contiguous)
  const int tid = threadIdx.x;
  const int row0 = blockIdx.x * MROWS;
  const int colbase = blockIdx.y * 128;

  const int lane = tid & 31;
  const int wid = tid >> 5;
  const int half_id = lane >> 4;
  const int l16 = lane & 15;

  v8f c[5] = {};
  for (int kb = 0; kb < K; kb += 128) {
    __syncthreads();
    // stage A chunk: MROWS rows x 128 k  (uint4 = 8 halves)
    for (int i = tid; i < MROWS * 16; i += 256) {
      int r = i >> 4, k8 = (i & 15) * 8;
      *(uint4*)(Ah + r * 128 + k8) =
          *(const uint4*)(A + (size_t)(row0 + r) * K + kb + k8);
    }
    // stage W chunk column-major: 128 cols x 128 k
    for (int i = tid; i < 128 * 16; i += 256) {
      int cc = i >> 4, k8 = (i & 15) * 8;
      *(uint4*)(Wcm + cc * 128 + k8) =
          *(const uint4*)(Wt + (size_t)(colbase + cc) * K + kb + k8);
    }
    __syncthreads();
#pragma unroll
    for (int kt = 0; kt < 4; ++kt) {
      // B 32x16 f16 fragment (contiguous k order chosen here): lanes 0-15 K=0..15, 16-31 K=16..31;
      // contiguous in column-major LDS -> 2x ds_load_b128
      v16h bfr = {};
      const _Float16* bp = Wcm + (wid * 16 + l16) * 128 + kt * 32 + 16 * half_id;
#pragma unroll
      for (int j = 0; j < 16; ++j) bfr[j] = bp[j];
#pragma unroll
      for (int rt = 0; rt < 5; ++rt) {
        // A 16x32 f16 fragment: K = j + 16*half -> one 16-half run (matches B)
        v16h a = {};
        const _Float16* ap = Ah + (rt * 16 + l16) * 128 + kt * 32 + 16 * half_id;
#pragma unroll
        for (int j = 0; j < 8; ++j) a[j] = ap[j];
#pragma unroll
        for (int j = 8; j < 16; ++j) a[j] = ap[j];
        c[rt] = __builtin_amdgcn_wmma_f32_16x16x32_f16(false, a, false, bfr, (short)0,
                                                       c[rt], false, false);
      }
    }
  }

  int col = colbase + wid * 16 + l16;
  float bv = bias ? bias[col] : 0.0f;
#pragma unroll
  for (int rt = 0; rt < 5; ++rt) {
#pragma unroll
    for (int r = 0; r < 8; ++r) {
      int row = row0 + rt * 16 + r + 8 * half_id; // C/D layout: VGPR r -> M=r (+8 upper)
      float val = c[rt][r] + bv;
      size_t idx = (size_t)row * NcTot + col;
      if (mode == 0) {
        Cacc[idx] += val;
      } else if (mode == 1) {
        Oh[idx] = (_Float16)val;
      } else {
        float gg = 0.5f * val * (1.0f + erff(val * 0.70710678118654752f));
        Oh[idx] = (_Float16)gg;
      }
    }
  }
}

// ---------------------------------------------------------------- attention edge ops
// eb[e][h] = sum_d ef[e][d]*We[d][h] + be[h]; one wave per edge, We in LDS.
__global__ void edge_bias_kernel(const float* __restrict__ ef, const float* __restrict__ We,
                                 const float* __restrict__ be, float* __restrict__ eb) {
  __shared__ float Ws[DD * HH];
  for (int i = threadIdx.x; i < DD * HH; i += blockDim.x) Ws[i] = We[i];
  __syncthreads();
  int wid = threadIdx.x >> 5, lane = threadIdx.x & 31;
  int e = blockIdx.x * 8 + wid;
  if (e >= EE) return;
  float4 v = *(const float4*)(ef + (size_t)e * DD + lane * 4);
  float vv[4] = {v.x, v.y, v.z, v.w};
  float acc[HH];
#pragma unroll
  for (int h = 0; h < HH; ++h) acc[h] = 0.0f;
#pragma unroll
  for (int j = 0; j < 4; ++j) {
    int d = lane * 4 + j;
#pragma unroll
    for (int h = 0; h < HH; ++h) acc[h] += vv[j] * Ws[d * HH + h];
  }
#pragma unroll
  for (int h = 0; h < HH; ++h) {
    float s = acc[h];
#pragma unroll
    for (int m2 = 16; m2 >= 1; m2 >>= 1) s += __shfl_xor(s, m2);
    if (lane == 0) eb[(size_t)e * HH + h] = s + be[h];
  }
}

// score[e][h] = 0.25 * dot16(k[src],q[dst]) + eb ; atomic segment max into m[dst][h]
__global__ void score_kernel(const _Float16* __restrict__ kh, const _Float16* __restrict__ qh,
                             const int* __restrict__ src, const int* __restrict__ dst,
                             const float* __restrict__ eb, float* __restrict__ score,
                             float* __restrict__ m) {
  int t = blockIdx.x * blockDim.x + threadIdx.x;
  int e = t >> 5;
  if (e >= EE) return;
  int lane = t & 31;
  int head = lane >> 2, q4 = lane & 3;
  int s = src[e], d = dst[e];
  int base = head * 16 + q4 * 4;
  const _Float16* kp = kh + (size_t)s * DD + base;
  const _Float16* qp = qh + (size_t)d * DD + base;
  float p = 0.0f;
#pragma unroll
  for (int j = 0; j < 4; ++j) p += (float)kp[j] * (float)qp[j];
  p += __shfl_xor(p, 1);
  p += __shfl_xor(p, 2);
  if (q4 == 0) {
    float sc = p * 0.25f + eb[(size_t)e * HH + head];
    score[(size_t)e * HH + head] = sc;
    atomicMaxF(&m[(size_t)d * HH + head], sc);
  }
}

// ex = exp(score - m[dst]); denom[dst] += ex (score overwritten with ex)
__global__ void exp_kernel(float* __restrict__ score, const float* __restrict__ m,
                           const int* __restrict__ dst, float* __restrict__ denom) {
  int t = blockIdx.x * blockDim.x + threadIdx.x;
  if (t >= EE * HH) return;
  int e = t >> 3, hh = t & 7;
  int d = dst[e];
  float ex = expf(score[t] - m[(size_t)d * HH + hh]);
  score[t] = ex;
  atomicAdd(&denom[(size_t)d * HH + hh], ex);
}

// attacc[dst][d] += v[src][d] * ex/denom ; wave per edge, lane = 4 dims (one head)
__global__ void agg_kernel(const _Float16* __restrict__ vh, const int* __restrict__ src,
                           const int* __restrict__ dst, const float* __restrict__ ex,
                           const float* __restrict__ denom, float* __restrict__ attacc) {
  int t = blockIdx.x * blockDim.x + threadIdx.x;
  int e = t >> 5;
  if (e >= EE) return;
  int lane = t & 31;
  int head = lane >> 2;
  int s = src[e], d = dst[e];
  float a = ex[(size_t)e * HH + head] /
            fmaxf(denom[(size_t)d * HH + head], 1e-9f);
  const _Float16* vp = vh + (size_t)s * DD + lane * 4;
  float* op = attacc + (size_t)d * DD + lane * 4;
#pragma unroll
  for (int j = 0; j < 4; ++j) atomicAdd(op + j, a * (float)vp[j]);
}

// ---------------------------------------------------------------- host orchestration
extern "C" void kernel_launch(void* const* d_in, const int* in_sizes, int n_in,
                              void* d_out, int out_size, void* d_ws, size_t ws_size,
                              hipStream_t stream) {
  (void)in_sizes; (void)n_in; (void)out_size; (void)ws_size;
  const float* h = (const float*)d_in[0];
  const int* srcp[2] = {(const int*)d_in[1], (const int*)d_in[4]};
  const int* dstp[2] = {(const int*)d_in[2], (const int*)d_in[5]};
  const float* efp[2] = {(const float*)d_in[3], (const float*)d_in[6]};
  const float* gcn_W = (const float*)d_in[7];  const float* gcn_b = (const float*)d_in[8];
  const float* Wq = (const float*)d_in[9];     const float* bq = (const float*)d_in[10];
  const float* Wk = (const float*)d_in[11];    const float* bk = (const float*)d_in[12];
  const float* Wv = (const float*)d_in[13];    const float* bv = (const float*)d_in[14];
  const float* We = (const float*)d_in[15];    const float* be = (const float*)d_in[16];
  const float* Wo = (const float*)d_in[17];    const float* bo = (const float*)d_in[18];
  const float* lnl_g = (const float*)d_in[19]; const float* lnl_b = (const float*)d_in[20];
  const float* lna_g = (const float*)d_in[21]; const float* lna_b = (const float*)d_in[22];
  const float* lnf_g = (const float*)d_in[23]; const float* lnf_b = (const float*)d_in[24];
  const float* W1 = (const float*)d_in[25];    const float* b1 = (const float*)d_in[26];
  const float* W2 = (const float*)d_in[27];    const float* b2 = (const float*)d_in[28];
  float* out = (float*)d_out;

  char* base = (char*)d_ws;
  size_t off = 0;
  auto take = [&](size_t bytes) -> void* {
    void* p = base + off;
    off += (bytes + 255) & ~size_t(255);
    return p;
  };
  const size_t ND = (size_t)NN * DD;
  // all weight buffers hold TRANSPOSED (Nc x K) f16 copies
  _Float16* gcnWh = (_Float16*)take(2 * DD * DD * 2);
  _Float16* Wqh = (_Float16*)take(2 * DD * DD * 2);
  _Float16* Wkh = (_Float16*)take(2 * DD * DD * 2);
  _Float16* Wvh = (_Float16*)take(2 * DD * DD * 2);
  _Float16* Woh = (_Float16*)take(2 * DD * DD * 2);
  _Float16* W1h = (_Float16*)take(DD * 256 * 2);
  _Float16* W2h = (_Float16*)take(256 * DD * 2);
  float* hl = (float*)take(ND * 4);          // reused as t1h (N*256 f16) at the end
  float* deg = (float*)take(4 * NN * 4);     // [e0_out | e0_in | e1_out | e1_in]
  float* h1 = (float*)take(ND * 4);
  float* agg = (float*)take(ND * 4);         // also attacc
  _Float16* xh = (_Float16*)take(ND * 2);    // also atth
  float* hc = (float*)take(ND * 4);
  _Float16* hch = (_Float16*)take(ND * 2);   // also hfh
  _Float16* qh = (_Float16*)take(ND * 2);
  _Float16* kh = (_Float16*)take(ND * 2);
  _Float16* vh = (_Float16*)take(ND * 2);
  float* eb[2];
  eb[0] = (float*)take((size_t)EE * HH * 4);
  eb[1] = (float*)take((size_t)EE * HH * 4);
  float* scoreB = (float*)take((size_t)EE * HH * 4);
  float* mB = (float*)take((size_t)NN * HH * 4);
  float* denomB = (float*)take((size_t)NN * HH * 4);
  _Float16* t1h = (_Float16*)hl; // N x 256 f16 fits in hl (N x 128 f32)

  auto cgrid = [](size_t n) { return dim3((unsigned)((n + 255) / 256)); };
  auto gemm = [&](const _Float16* A, int K, const _Float16* Wm, int Nc,
                  const float* bias, float* Cacc, _Float16* Oh, int mode) {
    dim3 grid(NN / MROWS, Nc / 128);
    size_t sh = (size_t)(MROWS * 128 + 128 * 128) * sizeof(_Float16);
    gemm_wmma_kernel<<<grid, 256, sh, stream>>>(A, K, Wm, Nc, bias, Cacc, Oh, mode);
  };

  // weights -> transposed f16
  for (int i = 0; i < 2; ++i) {
    size_t wo = (size_t)i * DD * DD;
    convT_f32_f16<<<cgrid(DD * DD), 256, 0, stream>>>(gcn_W + wo, gcnWh + wo, DD, DD);
    convT_f32_f16<<<cgrid(DD * DD), 256, 0, stream>>>(Wq + wo, Wqh + wo, DD, DD);
    convT_f32_f16<<<cgrid(DD * DD), 256, 0, stream>>>(Wk + wo, Wkh + wo, DD, DD);
    convT_f32_f16<<<cgrid(DD * DD), 256, 0, stream>>>(Wv + wo, Wvh + wo, DD, DD);
    convT_f32_f16<<<cgrid(DD * DD), 256, 0, stream>>>(Wo + wo, Woh + wo, DD, DD);
  }
  convT_f32_f16<<<cgrid(DD * 256), 256, 0, stream>>>(W1, W1h, DD, 256);
  convT_f32_f16<<<cgrid(256 * DD), 256, 0, stream>>>(W2, W2h, 256, DD);

  // local GraphConv block: h1 = h + sum_i gcn_i(layernorm(h))
  ln_kernel<<<NN / 8, 256, 0, stream>>>(h, lnl_g, lnl_b, hl, nullptr);
  fill_f32<<<cgrid(4 * NN), 256, 0, stream>>>(deg, 0.0f, 4 * NN);
  deg_accum<<<cgrid(EE), 256, 0, stream>>>(srcp[0], dstp[0], deg, deg + NN);
  deg_accum<<<cgrid(EE), 256, 0, stream>>>(srcp[1], dstp[1], deg + 2 * NN, deg + 3 * NN);
  rs_finalize<<<cgrid(4 * NN), 256, 0, stream>>>(deg, 4 * NN);
  copy_f32<<<cgrid(ND), 256, 0, stream>>>(h, h1, (int)ND);
  for (int i = 0; i < 2; ++i) {
    float* rs_out = deg + 2 * i * NN;
    float* rs_in = deg + (2 * i + 1) * NN;
    fill_f32<<<cgrid(ND), 256, 0, stream>>>(agg, 0.0f, (int)ND);
    gcn_scatter<<<cgrid((size_t)EE * 32), 256, 0, stream>>>(hl, srcp[i], dstp[i], rs_out, agg);
    scale_rows_f16<<<cgrid(ND), 256, 0, stream>>>(agg, rs_in, xh);
    gemm(xh, DD, gcnWh + (size_t)i * DD * DD, DD, gcn_b + i * DD, h1, nullptr, 0);
  }

  // attention pre-norm + hop-invariant edge biases
  ln_kernel<<<NN / 8, 256, 0, stream>>>(h1, lna_g, lna_b, hc, nullptr);
  edge_bias_kernel<<<EE / 8, 256, 0, stream>>>(efp[0], We, be, eb[0]);
  edge_bias_kernel<<<EE / 8, 256, 0, stream>>>(efp[1], We + DD * HH, be + HH, eb[1]);

  // 2 hops of sparse multi-head attention; hc += sum_i attn_i(hc_snapshot)
  for (int hop = 0; hop < 2; ++hop) {
    conv_f32_f16<<<cgrid(ND), 256, 0, stream>>>(hc, hch, (int)ND);
    for (int i = 0; i < 2; ++i) {
      size_t wo = (size_t)i * DD * DD;
      gemm(hch, DD, Wqh + wo, DD, bq + i * DD, nullptr, qh, 1);
      gemm(hch, DD, Wkh + wo, DD, bk + i * DD, nullptr, kh, 1);
      gemm(hch, DD, Wvh + wo, DD, bv + i * DD, nullptr, vh, 1);
      fill_f32<<<cgrid((size_t)NN * HH), 256, 0, stream>>>(mB, -1e30f, NN * HH);
      fill_f32<<<cgrid((size_t)NN * HH), 256, 0, stream>>>(denomB, 0.0f, NN * HH);
      score_kernel<<<cgrid((size_t)EE * 32), 256, 0, stream>>>(kh, qh, srcp[i], dstp[i],
                                                               eb[i], scoreB, mB);
      exp_kernel<<<cgrid((size_t)EE * HH), 256, 0, stream>>>(scoreB, mB, dstp[i], denomB);
      fill_f32<<<cgrid(ND), 256, 0, stream>>>(agg, 0.0f, (int)ND);
      agg_kernel<<<cgrid((size_t)EE * 32), 256, 0, stream>>>(vh, srcp[i], dstp[i], scoreB,
                                                             denomB, agg);
      conv_f32_f16<<<cgrid(ND), 256, 0, stream>>>(agg, xh, (int)ND);
      gemm(xh, DD, Woh + wo, DD, bo + i * DD, hc, nullptr, 0);
    }
  }

  // FFN: out = hc + gelu(ln(hc)@W1+b1)@W2+b2
  ln_kernel<<<NN / 8, 256, 0, stream>>>(hc, lnf_g, lnf_b, nullptr, hch);
  gemm(hch, DD, W1h, 256, b1, nullptr, t1h, 2);
  copy_f32<<<cgrid(ND), 256, 0, stream>>>(hc, out, (int)ND);
  gemm(t1h, 256, W2h, DD, b2, out, nullptr, 0);
}
